// CrossBlock_5909875000097
// MI455X (gfx1250) — matrix operand order repaired and checked by reading, not compiled, not run on hardware
//
#include <hip/hip_runtime.h>
#include <hip/hip_bf16.h>

// ---- problem constants -----------------------------------------------------
#define Bb   4
#define Nn   1024
#define Dd   1024
#define Hh   16
#define DHh  64
#define D2   2048
#define BN   4096      // B*N tokens per stream
#define MT   8192      // rows for both streams concatenated
#define LDA  40        // LDS row stride in halfs: 32 K + 8 pad (16B aligned, conflict-free)
#define QK_SCALE 0.3535533905932738f   // DH^-0.25

typedef __attribute__((ext_vector_type(16))) _Float16 v16h;
typedef __attribute__((ext_vector_type(8)))  float    v8f;

// ---- optional gfx1250 async global->LDS path (ASYNCcnt) --------------------
#if defined(__has_builtin)
#if __has_builtin(__builtin_amdgcn_global_load_async_to_lds_b64) && \
    __has_builtin(__builtin_amdgcn_s_wait_asynccnt)
#define HAVE_ASYNC_LDS 1
#endif
#endif
#ifndef HAVE_ASYNC_LDS
#define HAVE_ASYNC_LDS 0
#endif

#if HAVE_ASYNC_LDS
// builtin is typed over 8-byte int vectors (per hipcc diagnostic):
//   (v2i addrspace(1)*, v2i addrspace(3)*, imm offset, imm cpol)
typedef __attribute__((__vector_size__(2 * sizeof(int)))) int v2i;
typedef __attribute__((address_space(1))) v2i g_v2i;
typedef __attribute__((address_space(3))) v2i l_v2i;
#endif

// Copy a 64-row x 32-half tile (rows contiguous in global, stride row_stride
// halfs) into a padded LDS tile (stride LDA halfs). Pure f16 copy, no
// conversion, so it is eligible for the async DMA path.
__device__ __forceinline__ void stage_rows_f16(_Float16* sdst,
                                               const _Float16* gsrc,
                                               size_t row_stride) {
  int tid = threadIdx.x;
#if HAVE_ASYNC_LDS
#pragma unroll
  for (int it = 0; it < 4; ++it) {
    int idx = tid + it * 128;          // 0..511 : 64 rows x 8 chunks of 8B
    int rr = idx >> 3, ch = idx & 7;
    const _Float16* g = gsrc + (size_t)rr * row_stride + ch * 4;
    _Float16* l = sdst + rr * LDA + ch * 4;
    __builtin_amdgcn_global_load_async_to_lds_b64((g_v2i*)g, (l_v2i*)l, 0, 0);
  }
#else
#pragma unroll
  for (int it = 0; it < 16; ++it) {
    int idx = tid + it * 128;
    int rr = idx >> 5, cc = idx & 31;
    sdst[rr * LDA + cc] = gsrc[(size_t)rr * row_stride + cc];
  }
#endif
}

__device__ __forceinline__ void stage_wait_async() {
#if HAVE_ASYNC_LDS
  __builtin_amdgcn_s_wait_asynccnt(0);
#endif
}

// ---- WMMA core helpers (layouts per CDNA5 ISA §7.12.2) ----------
__device__ __forceinline__ v8f wmma16(v16h a, v16h b, v8f c) {
  return __builtin_amdgcn_wmma_f32_16x16x32_f16(false, a, false, b, (short)0, c,
                                                false, false);
}

// A 16x32 f16 fragment. lanes 0-15: M=lane, K = {h*8..h*8+7, 16+h*8..16+h*8+7}
__device__ __forceinline__ v16h load_frag_a(const _Float16* base) {
  int lane = threadIdx.x & 31;
  int m = lane & 15, hh = lane >> 4;
  const _Float16* row = base + m * LDA;
  v16h f;
#pragma unroll
  for (int e = 0; e < 8; ++e) f[e] = row[hh * 8 + e];
#pragma unroll
  for (int e = 0; e < 8; ++e) f[8 + e] = row[16 + hh * 8 + e];
  return f;
}

// B 32x16 f16 fragment from an N-major LDS tile (tile[n][k]).
// lanes 0-15: N=lane, K = {h*8..h*8+7, 16+h*8..16+h*8+7} (h = lane>>4)
__device__ __forceinline__ v16h load_frag_b(const _Float16* base) {
  int lane = threadIdx.x & 31;
  int n = lane & 15, hh = lane >> 4;
  const _Float16* row = base + n * LDA;
  v16h f;
#pragma unroll
  for (int e = 0; e < 8; ++e) f[e] = row[hh * 8 + e];
#pragma unroll
  for (int e = 0; e < 8; ++e) f[8 + e] = row[16 + hh * 8 + e];
  return f;
}

__device__ __forceinline__ void zero8(v8f& c) {
#pragma unroll
  for (int r = 0; r < 8; ++r) c[r] = 0.0f;
}

// 64x64 macro tile, 4 waves; wave w -> quadrant (wr=w>>1, wc=w&1), 2x2 WMMAs.
__device__ __forceinline__ void mma_tile(const _Float16* sA, const _Float16* sB,
                                         v8f& c00, v8f& c01, v8f& c10, v8f& c11) {
  int w  = (int)(threadIdx.x >> 5);
  int wr = (w >> 1) & 1, wc = w & 1;
  v16h a0 = load_frag_a(sA + (wr * 32 +  0) * LDA);
  v16h a1 = load_frag_a(sA + (wr * 32 + 16) * LDA);
  v16h b0 = load_frag_b(sB + (wc * 32 +  0) * LDA);
  v16h b1 = load_frag_b(sB + (wc * 32 + 16) * LDA);
  c00 = wmma16(a0, b0, c00);
  c01 = wmma16(a0, b1, c01);
  c10 = wmma16(a1, b0, c10);
  c11 = wmma16(a1, b1, c11);
}

// C/D 16x16 f32: VGPR r holds row r (lanes 0-15) / row r+8 (lanes 16-31), col=lane&15
__device__ __forceinline__ int acc_row(int sub_i, int r) {
  int lane = threadIdx.x & 31;
  int wr = ((int)(threadIdx.x >> 5) >> 1) & 1;
  return wr * 32 + sub_i * 16 + r + ((lane >> 4) << 3);
}
__device__ __forceinline__ int acc_col(int sub_j) {
  int lane = threadIdx.x & 31;
  int wc = (int)(threadIdx.x >> 5) & 1;
  return wc * 32 + sub_j * 16 + (lane & 15);
}

#define PICK_ACC(i, j) ((i) == 0 ? ((j) == 0 ? c00 : c01) : ((j) == 0 ? c10 : c11))

// ---- K1: QK/V projection: f32 X @ f32 W + b, scaled, f16 out ---------------
__global__ void __launch_bounds__(128)
k_proj(const float* __restrict__ x0, const float* __restrict__ x1,
       const float* __restrict__ W, const float* __restrict__ bias,
       _Float16* __restrict__ out, float scale)
{
  __shared__ __align__(16) _Float16 sA[64 * LDA];
  __shared__ __align__(16) _Float16 sB[64 * LDA];
  const int tid  = threadIdx.x;
  const int col0 = blockIdx.x * 64;
  const int row0 = blockIdx.y * 64;
  const float* xb = (row0 < BN) ? x0 : x1;
  const int    rb = (row0 < BN) ? row0 : row0 - BN;

  v8f c00, c01, c10, c11;
  zero8(c00); zero8(c01); zero8(c10); zero8(c11);

  for (int kk = 0; kk < Dd; kk += 32) {
#pragma unroll
    for (int it = 0; it < 16; ++it) {
      int idx = tid + it * 128;
      int rr = idx >> 5, cc = idx & 31;
      sA[rr * LDA + cc] = (_Float16)xb[(size_t)(rb + rr) * Dd + kk + cc];
      int nn = idx & 63, k2 = idx >> 6;
      sB[nn * LDA + k2] = (_Float16)W[(size_t)(kk + k2) * Dd + col0 + nn];
    }
    __syncthreads();
    mma_tile(sA, sB, c00, c01, c10, c11);
    __syncthreads();
  }

#pragma unroll
  for (int i = 0; i < 2; ++i)
#pragma unroll
    for (int j = 0; j < 2; ++j) {
      v8f c = PICK_ACC(i, j);
      int gc = col0 + acc_col(j);
      float bb = bias[gc];
#pragma unroll
      for (int r = 0; r < 8; ++r) {
        int gr = row0 + acc_row(i, r);
        out[(size_t)gr * Dd + gc] = (_Float16)((c[r] + bb) * scale);
      }
    }
}

// ---- K2: per-(b,h) sim = qk0 @ qk1^T (K = DH = 64) -------------------------
__global__ void __launch_bounds__(128)
k_sim(const _Float16* __restrict__ qk, float* __restrict__ sim)
{
  __shared__ __align__(16) _Float16 sA[64 * LDA];
  __shared__ __align__(16) _Float16 sB[64 * LDA];
  const int z = blockIdx.z, b = z >> 4, h = z & 15;
  const int col0 = blockIdx.x * 64, row0 = blockIdx.y * 64;
  const _Float16* q0 = qk + (size_t)(b * Nn) * Dd + h * DHh;
  const _Float16* q1 = qk + (size_t)(BN + b * Nn) * Dd + h * DHh;

  v8f c00, c01, c10, c11;
  zero8(c00); zero8(c01); zero8(c10); zero8(c11);

  for (int kk = 0; kk < DHh; kk += 32) {
    // pure f16 row copies -> async DMA eligible.
    // qk1 rows ARE the N dimension of B => N-major stage is a row copy.
    stage_rows_f16(sA, q0 + (size_t)row0 * Dd + kk, Dd);
    stage_rows_f16(sB, q1 + (size_t)col0 * Dd + kk, Dd);
    stage_wait_async();
    __syncthreads();
    mma_tile(sA, sB, c00, c01, c10, c11);
    __syncthreads();
  }

  float* simz = sim + (size_t)z * Nn * Nn;
#pragma unroll
  for (int i = 0; i < 2; ++i)
#pragma unroll
    for (int j = 0; j < 2; ++j) {
      v8f c = PICK_ACC(i, j);
      int gc = col0 + acc_col(j);
#pragma unroll
      for (int r = 0; r < 8; ++r) {
        int gr = row0 + acc_row(i, r);
        simz[(size_t)gr * Nn + gc] = c[r];
      }
    }
}

// ---- K3a: softmax row stats (axis -1): one wave per row --------------------
__global__ void __launch_bounds__(256)
k_rowstats(const float* __restrict__ sim, float* __restrict__ rmax,
           float* __restrict__ rsum)
{
  int wid = threadIdx.x >> 5, lane = threadIdx.x & 31;
  int R = blockIdx.x * 8 + wid;               // R = bh*1024 + i
  const float* row = sim + (size_t)R * Nn;
  float m = -1e30f, s = 0.0f;
#pragma unroll 4
  for (int t = 0; t < 32; ++t) {
    float x = row[lane + 32 * t];
    float nm = fmaxf(m, x);
    s = s * __expf(m - nm) + __expf(x - nm);
    m = nm;
  }
  for (int off = 16; off > 0; off >>= 1) {
    float om = __shfl_xor(m, off, 32);
    float os = __shfl_xor(s, off, 32);
    float nm = fmaxf(m, om);
    s = s * __expf(m - nm) + os * __expf(om - nm);
    m = nm;
  }
  if (lane == 0) { rmax[R] = m; rsum[R] = s; }
}

// ---- K3b: softmax column stats (axis -2): thread-per-column ----------------
__global__ void __launch_bounds__(256)
k_colstats(const float* __restrict__ sim, float* __restrict__ cmax,
           float* __restrict__ csum)
{
  int bh = blockIdx.x >> 2;
  int c  = ((blockIdx.x & 3) << 8) + threadIdx.x;
  const float* base = sim + (size_t)bh * Nn * Nn + c;
  float m = -1e30f, s = 0.0f;
  for (int j = 0; j < Nn; ++j) {
    float x = base[(size_t)j * Nn];
    float nm = fmaxf(m, x);
    s = s * __expf(m - nm) + __expf(x - nm);
    m = nm;
  }
  cmax[bh * Nn + c] = m;
  csum[bh * Nn + c] = s;
}

// ---- K4/K5: m = softmax(sim) @ V  (mode 0: attn01@v1 ; mode 1: attn10^T@v0)
__global__ void __launch_bounds__(128)
k_attnv(const _Float16* __restrict__ vh, const float* __restrict__ sim,
        const float* __restrict__ pmax, const float* __restrict__ psum,
        _Float16* __restrict__ mout, int mode)
{
  __shared__ __align__(16) _Float16 sA[64 * LDA];
  __shared__ __align__(16) _Float16 sB[64 * LDA];
  __shared__ float rm[64], rs[64];
  const int tid = threadIdx.x;
  const int z = blockIdx.z, b = z >> 4, h = z & 15;
  const int row0 = blockIdx.y * 64;
  const float* simz = sim + (size_t)z * Nn * Nn;
  const _Float16* vsrc = vh + (mode ? (size_t)0 : (size_t)BN * Dd);  // v1 / v0
  _Float16* dst = mout + (mode ? (size_t)BN * Dd : (size_t)0);

  if (tid < 64) {
    rm[tid] = pmax[z * Nn + row0 + tid];
    rs[tid] = 1.0f / psum[z * Nn + row0 + tid];
  }
  __syncthreads();

  v8f c00, c01, c10, c11;
  zero8(c00); zero8(c01); zero8(c10); zero8(c11);

  for (int kk = 0; kk < Nn; kk += 32) {
#pragma unroll
    for (int it = 0; it < 16; ++it) {
      int idx = tid + it * 128;
      if (mode == 0) {          // A[m][k] = attn01[row0+m][kk+k]
        int rr = idx >> 5, cc = idx & 31;
        float x = simz[(size_t)(row0 + rr) * Nn + kk + cc];
        sA[rr * LDA + cc] = (_Float16)(__expf(x - rm[rr]) * rs[rr]);
      } else {                  // A[m][k] = attn10[kk+k][row0+m]
        int mm = idx & 63, k2 = idx >> 6;
        float x = simz[(size_t)(kk + k2) * Nn + row0 + mm];
        sA[mm * LDA + k2] = (_Float16)(__expf(x - rm[mm]) * rs[mm]);
      }
      int nn = idx & 63, k2 = idx >> 6;  // B[k][n] = v[kk+k][h*64+n], N-major stage
      sB[nn * LDA + k2] = vsrc[(size_t)(b * Nn + kk + k2) * Dd + h * DHh + nn];
    }
    __syncthreads();
    mma_tile(sA, sB, c00, c01, c10, c11);
    __syncthreads();
  }

#pragma unroll
  for (int i = 0; i < 2; ++i)
#pragma unroll
    for (int j = 0; j < 2; ++j) {
      v8f c = PICK_ACC(i, j);
      int gc = acc_col(j);   // 0..63 == DH
#pragma unroll
      for (int r = 0; r < 8; ++r) {
        int gr = row0 + acc_row(i, r);
        dst[(size_t)(b * Nn + gr) * Dd + h * DHh + gc] = (_Float16)c[r];
      }
    }
}

// ---- K6: generic f16A @ f32W + bias -> f16 out (used for Wo) ---------------
__global__ void __launch_bounds__(128)
k_gemm_h(const _Float16* __restrict__ A, const float* __restrict__ W,
         const float* __restrict__ bias, _Float16* __restrict__ out,
         int K, int Nout)
{
  __shared__ __align__(16) _Float16 sA[64 * LDA];
  __shared__ __align__(16) _Float16 sB[64 * LDA];
  const int tid = threadIdx.x;
  const int col0 = blockIdx.x * 64, row0 = blockIdx.y * 64;

  v8f c00, c01, c10, c11;
  zero8(c00); zero8(c01); zero8(c10); zero8(c11);

  for (int kk = 0; kk < K; kk += 32) {
    stage_rows_f16(sA, A + (size_t)row0 * K + kk, K);   // async-eligible
#pragma unroll
    for (int it = 0; it < 16; ++it) {
      int idx = tid + it * 128;
      int nn = idx & 63, k2 = idx >> 6;
      sB[nn * LDA + k2] = (_Float16)W[(size_t)(kk + k2) * Nout + col0 + nn];
    }
    stage_wait_async();
    __syncthreads();
    mma_tile(sA, sB, c00, c01, c10, c11);
    __syncthreads();
  }

#pragma unroll
  for (int i = 0; i < 2; ++i)
#pragma unroll
    for (int j = 0; j < 2; ++j) {
      v8f c = PICK_ACC(i, j);
      int gc = col0 + acc_col(j);
      float bb = bias[gc];
#pragma unroll
      for (int r = 0; r < 8; ++r) {
        int gr = row0 + acc_row(i, r);
        out[(size_t)gr * Nout + gc] = (_Float16)(c[r] + bb);
      }
    }
}

// ---- K7: FFN1: [x | mo] @ Wf1 + bf1 -> f32 (concat fused in A-stage) -------
__global__ void __launch_bounds__(128)
k_ffn1(const float* __restrict__ x0, const float* __restrict__ x1,
       const _Float16* __restrict__ mo, const float* __restrict__ Wf1,
       const float* __restrict__ bf1, float* __restrict__ hbuf)
{
  __shared__ __align__(16) _Float16 sA[64 * LDA];
  __shared__ __align__(16) _Float16 sB[64 * LDA];
  const int tid = threadIdx.x;
  const int col0 = blockIdx.x * 64, row0 = blockIdx.y * 64;
  const float* xb = (row0 < BN) ? x0 : x1;
  const int    rb = (row0 < BN) ? row0 : row0 - BN;

  v8f c00, c01, c10, c11;
  zero8(c00); zero8(c01); zero8(c10); zero8(c11);

  for (int kk = 0; kk < D2; kk += 32) {
    if (kk < Dd) {
#pragma unroll
      for (int it = 0; it < 16; ++it) {
        int idx = tid + it * 128;
        int rr = idx >> 5, cc = idx & 31;
        sA[rr * LDA + cc] = (_Float16)xb[(size_t)(rb + rr) * Dd + kk + cc];
      }
    } else {
      stage_rows_f16(sA, mo + (size_t)row0 * Dd + (kk - Dd), Dd);  // async-eligible
    }
#pragma unroll
    for (int it = 0; it < 16; ++it) {
      int idx = tid + it * 128;
      int nn = idx & 63, k2 = idx >> 6;
      sB[nn * LDA + k2] = (_Float16)Wf1[(size_t)(kk + k2) * D2 + col0 + nn];
    }
    // prefetch next W tile one K-step ahead (lowers to global_prefetch_b8)
    if (kk + 32 < D2)
      __builtin_prefetch(&Wf1[(size_t)(kk + 32 + (tid >> 6)) * D2 + col0 + (tid & 63)], 0, 1);
    stage_wait_async();
    __syncthreads();
    mma_tile(sA, sB, c00, c01, c10, c11);
    __syncthreads();
  }

#pragma unroll
  for (int i = 0; i < 2; ++i)
#pragma unroll
    for (int j = 0; j < 2; ++j) {
      v8f c = PICK_ACC(i, j);
      int gc = col0 + acc_col(j);
      float bb = bf1[gc];
#pragma unroll
      for (int r = 0; r < 8; ++r) {
        int gr = row0 + acc_row(i, r);
        hbuf[(size_t)gr * D2 + gc] = c[r] + bb;
      }
    }
}

// ---- K8: LayerNorm + exact GELU, one block per row of 2048 -----------------
__global__ void __launch_bounds__(256)
k_lngelu(const float* __restrict__ hbuf, const float* __restrict__ gamma,
         const float* __restrict__ beta, _Float16* __restrict__ gbuf)
{
  __shared__ float s1[256], s2[256];
  const int tid = threadIdx.x;
  const float* row = hbuf + (size_t)blockIdx.x * D2;
  float a = 0.0f, b = 0.0f;
#pragma unroll
  for (int i = 0; i < 8; ++i) {
    float v = row[tid + 256 * i];
    a += v; b += v * v;
  }
  s1[tid] = a; s2[tid] = b;
  __syncthreads();
  for (int s = 128; s > 0; s >>= 1) {
    if (tid < s) { s1[tid] += s1[tid + s]; s2[tid] += s2[tid + s]; }
    __syncthreads();
  }
  float mu  = s1[0] * (1.0f / D2);
  float var = s2[0] * (1.0f / D2) - mu * mu;
  float rstd = rsqrtf(var + 1e-5f);
  _Float16* orow = gbuf + (size_t)blockIdx.x * D2;
#pragma unroll
  for (int i = 0; i < 8; ++i) {
    int c = tid + 256 * i;
    float v = (row[c] - mu) * rstd * gamma[c] + beta[c];
    float g = 0.5f * v * (1.0f + erff(v * 0.70710678118654752f));
    orow[c] = (_Float16)g;
  }
}

// ---- K9: FFN2 + bias + residual -> d_out -----------------------------------
__global__ void __launch_bounds__(128)
k_ffn2(const _Float16* __restrict__ gbuf, const float* __restrict__ Wf2,
       const float* __restrict__ bf2, const float* __restrict__ x0,
       const float* __restrict__ x1, float* __restrict__ y)
{
  __shared__ __align__(16) _Float16 sA[64 * LDA];
  __shared__ __align__(16) _Float16 sB[64 * LDA];
  const int tid = threadIdx.x;
  const int col0 = blockIdx.x * 64, row0 = blockIdx.y * 64;
  const float* xb = (row0 < BN) ? x0 : x1;
  const int    rb = (row0 < BN) ? row0 : row0 - BN;

  v8f c00, c01, c10, c11;
  zero8(c00); zero8(c01); zero8(c10); zero8(c11);

  for (int kk = 0; kk < D2; kk += 32) {
    stage_rows_f16(sA, gbuf + (size_t)row0 * D2 + kk, D2);   // async-eligible
#pragma unroll
    for (int it = 0; it < 16; ++it) {
      int idx = tid + it * 128;
      int nn = idx & 63, k2 = idx >> 6;
      sB[nn * LDA + k2] = (_Float16)Wf2[(size_t)(kk + k2) * Dd + col0 + nn];
    }
    stage_wait_async();
    __syncthreads();
    mma_tile(sA, sB, c00, c01, c10, c11);
    __syncthreads();
  }

#pragma unroll
  for (int i = 0; i < 2; ++i)
#pragma unroll
    for (int j = 0; j < 2; ++j) {
      v8f c = PICK_ACC(i, j);
      int gc = col0 + acc_col(j);
      float bb = bf2[gc];
#pragma unroll
      for (int r = 0; r < 8; ++r) {
        int gr = row0 + acc_row(i, r);
        float res = xb[(size_t)(rb + acc_row(i, r)) * Dd + gc];
        y[(size_t)gr * Dd + gc] = c[r] + bb + res;
      }
    }
}

// ---- host orchestration ----------------------------------------------------
extern "C" void kernel_launch(void* const* d_in, const int* in_sizes, int n_in,
                              void* d_out, int out_size, void* d_ws, size_t ws_size,
                              hipStream_t stream) {
  const float* x0    = (const float*)d_in[0];
  const float* x1    = (const float*)d_in[1];
  const float* Wqk   = (const float*)d_in[2];
  const float* bqk   = (const float*)d_in[3];
  const float* Wv    = (const float*)d_in[4];
  const float* bv    = (const float*)d_in[5];
  const float* Wo    = (const float*)d_in[6];
  const float* bo    = (const float*)d_in[7];
  const float* Wf1   = (const float*)d_in[8];
  const float* bf1   = (const float*)d_in[9];
  const float* gamma = (const float*)d_in[10];
  const float* beta  = (const float*)d_in[11];
  const float* Wf2   = (const float*)d_in[12];
  const float* bf2   = (const float*)d_in[13];
  float* y = (float*)d_out;

  char* ws = (char*)d_ws;
  const size_t MB = 1ull << 20;
  _Float16* qk = (_Float16*)(ws);             // 16 MB  [8192][1024] f16
  _Float16* vv = (_Float16*)(ws + 16 * MB);   // 16 MB
  _Float16* mb = (_Float16*)(ws + 32 * MB);   // 16 MB  merged attn out
  _Float16* mo = (_Float16*)(ws + 48 * MB);   // 16 MB  after Wo
  float* rmax  = (float*)(ws + 64 * MB);      // 4 x 256 KB stats
  float* rsum  = rmax + 64 * Nn;
  float* cmax  = rsum + 64 * Nn;
  float* csum  = cmax + 64 * Nn;
  float* simb  = (float*)(ws + 66 * MB);      // 256 MB [64][1024][1024] f32
  float* hbuf  = simb;                        // reuse sim region after attn
  _Float16* gbuf = (_Float16*)(ws + 130 * MB);// 32 MB, still inside dead sim region

  dim3 blk(128);

  // projections (QK scaled by DH^-0.25 so qk0.qk1 carries DH^-0.5)
  k_proj<<<dim3(16, 128), blk, 0, stream>>>(x0, x1, Wqk, bqk, qk, QK_SCALE);
  k_proj<<<dim3(16, 128), blk, 0, stream>>>(x0, x1, Wv,  bv,  vv, 1.0f);

  // similarity + bidirectional softmax stats
  k_sim<<<dim3(16, 16, 64), blk, 0, stream>>>(qk, simb);
  k_rowstats<<<dim3(8192), dim3(256), 0, stream>>>(simb, rmax, rsum);
  k_colstats<<<dim3(256),  dim3(256), 0, stream>>>(simb, cmax, csum);

  // m0 = softmax(sim,-1) @ v1 ; m1 = softmax(sim,-2)^T @ v0
  k_attnv<<<dim3(1, 16, 64), blk, 0, stream>>>(vv, simb, rmax, rsum, mb, 0);
  k_attnv<<<dim3(1, 16, 64), blk, 0, stream>>>(vv, simb, cmax, csum, mb, 1);

  // output projection
  k_gemm_h<<<dim3(16, 128), blk, 0, stream>>>(mb, Wo, bo, mo, Dd, Dd);

  // FFN
  k_ffn1<<<dim3(32, 128), blk, 0, stream>>>(x0, x1, mo, Wf1, bf1, hbuf);
  k_lngelu<<<dim3(8192), dim3(256), 0, stream>>>(hbuf, gamma, beta, gbuf);
  k_ffn2<<<dim3(16, 128), blk, 0, stream>>>(gbuf, Wf2, bf2, x0, x1, y);

  (void)in_sizes; (void)n_in; (void)out_size; (void)ws_size;
}
